// TrajectoryImitationKinematicConstrainedCNNLSTM_69380901700103
// MI455X (gfx1250) — compile-verified, matrix operand order checked by reading, not run on hardware
//
#include <hip/hip_runtime.h>
#include <hip/hip_bf16.h>

// ---------------------------------------------------------------------------
// gfx1250 WMMA types
// ---------------------------------------------------------------------------
typedef __attribute__((ext_vector_type(16))) _Float16 v16h;
typedef __attribute__((ext_vector_type(8)))  float    v8f;

union V16H { v16h v; _Float16 e[16]; };
union V8F  { v8f  v; float    e[8];  };

#define WMMA_F16(c, a, b) __builtin_amdgcn_wmma_f32_16x16x32_f16( \
        false, (a), false, (b), (short)0, (c), false, false)

// ---------------------------------------------------------------------------
// Generic WMMA GEMM:  Out(M x N) = act( scale[m]*(A(MxK)*B(KxN)) + bias[m]
//                                       + residual )
// A row-major (weights).  B: transB==0 -> B[k*N+n], transB==1 -> B[n*K+k].
// Each wave owns a 32(M) x 16(N) tile: two A-row fragments share one B
// fragment, so every B load feeds two v_wmma (B is the streamed operand).
// K swept 32 (or 64 with four independent accumulator chains) per step,
// on-the-fly f32->f16 conversion, f32 WMMA accumulation.
// Interior tiles are fully unguarded; edge tiles execute clamped-address
// loads UNCONDITIONALLY (always in-bounds) and select zero afterwards, so
// they lower to v_cndmask instead of per-element saveexec branches.
// Tile index is wave-uniform so EXEC is all-1s at every WMMA issue.
// blockIdx.y = batch image (strideB/strideC/strideR elements apart).
// ---------------------------------------------------------------------------
__global__ void __launch_bounds__(256)
gemm_wmma_kernel(const float* __restrict__ A, const float* __restrict__ B,
                 float* __restrict__ C,
                 int M, int N, int K, int transB,
                 const float* __restrict__ scale, const float* __restrict__ bias,
                 const float* __restrict__ residual, int act,
                 long strideB, long strideC, long strideR)
{
    const int wave = threadIdx.x >> 5;      // 8 waves / block (wave32)
    const int lane = threadIdx.x & 31;
    const int nTilesN = (N + 15) >> 4;
    const int nTilesM = (M + 31) >> 5;      // 32 output rows per wave
    const int tile = blockIdx.x * 8 + wave; // wave-uniform
    if (tile >= nTilesM * nTilesN) return;

    const int mt = (tile / nTilesN) << 5;
    const int nt = (tile % nTilesN) << 4;
    const float* __restrict__ Bp = B + (long)blockIdx.y * strideB;
    float*       __restrict__ Cp = C + (long)blockIdx.y * strideC;

    const int half  = lane >> 4;    // fragment K-half (ISA 7.12.2)
    const int ml    = lane & 15;    // m (A) / n (B,D) within tile
    const int kbase = half << 3;    // 0 or 8

    const bool interior = (mt + 32 <= M) && (nt + 16 <= N);
    const int  Kmain    = K & ~31;

    v8f acc0 = {};                  // rows mt .. mt+15
    v8f acc1 = {};                  // rows mt+16 .. mt+31

    if (interior) {
        // ------------------------- fast path -------------------------------
        v8f acc0b = {}, acc1b = {};
        const float* __restrict__ Ap0 = A + (long)(mt      + ml) * K + kbase;
        const float* __restrict__ Ap1 = A + (long)(mt + 16 + ml) * K + kbase;
        int k0 = 0;
        if (transB) {
            const float* __restrict__ Bq = Bp + (long)(nt + ml) * K + kbase;
            for (; k0 + 64 <= K; k0 += 64) {   // 4 independent acc chains
                V16H a0, a1, b0, a2, a3, b1;
#pragma unroll
                for (int j = 0; j < 16; ++j) {
                    const int o = ((j >> 3) << 4) + (j & 7);
                    a0.e[j] = (_Float16)Ap0[k0 + o];
                    a1.e[j] = (_Float16)Ap1[k0 + o];
                    b0.e[j] = (_Float16)Bq [k0 + o];
                    a2.e[j] = (_Float16)Ap0[k0 + 32 + o];
                    a3.e[j] = (_Float16)Ap1[k0 + 32 + o];
                    b1.e[j] = (_Float16)Bq [k0 + 32 + o];
                }
                acc0  = WMMA_F16(acc0,  a0.v, b0.v);
                acc1  = WMMA_F16(acc1,  a1.v, b0.v);
                acc0b = WMMA_F16(acc0b, a2.v, b1.v);
                acc1b = WMMA_F16(acc1b, a3.v, b1.v);
            }
            for (; k0 < Kmain; k0 += 32) {
                V16H a0, a1, b0;
#pragma unroll
                for (int j = 0; j < 16; ++j) {
                    const int o = ((j >> 3) << 4) + (j & 7);
                    a0.e[j] = (_Float16)Ap0[k0 + o];
                    a1.e[j] = (_Float16)Ap1[k0 + o];
                    b0.e[j] = (_Float16)Bq [k0 + o];
                }
                acc0 = WMMA_F16(acc0, a0.v, b0.v);
                acc1 = WMMA_F16(acc1, a1.v, b0.v);
            }
            if (k0 < K) {    // ragged K tail, branch-free selects
                V16H a0, a1, b0;
#pragma unroll
                for (int j = 0; j < 16; ++j) {
                    const int k  = k0 + ((j >> 3) << 4) + kbase + (j & 7);
                    const int kc = k < K ? k : K - 1;
                    const float av0 = A[(long)(mt      + ml) * K + kc];
                    const float av1 = A[(long)(mt + 16 + ml) * K + kc];
                    const float bv  = Bp[(long)(nt + ml) * K + kc];
                    a0.e[j] = (_Float16)(k < K ? av0 : 0.0f);
                    a1.e[j] = (_Float16)(k < K ? av1 : 0.0f);
                    b0.e[j] = (_Float16)(k < K ? bv  : 0.0f);
                }
                acc0 = WMMA_F16(acc0, a0.v, b0.v);
                acc1 = WMMA_F16(acc1, a1.v, b0.v);
            }
        } else {
            // Lanes 0..15 read 16 consecutive n for each k: coalesced.
            const float* __restrict__ Bg = Bp + (long)kbase * N + nt + ml;
            for (; k0 + 64 <= K; k0 += 64) {   // 4 independent acc chains
                V16H a0, a1, b0, a2, a3, b1;
#pragma unroll
                for (int j = 0; j < 16; ++j) {
                    const int o = ((j >> 3) << 4) + (j & 7);
                    a0.e[j] = (_Float16)Ap0[k0 + o];
                    a1.e[j] = (_Float16)Ap1[k0 + o];
                    b0.e[j] = (_Float16)Bg[(long)(k0 + o) * N];
                    a2.e[j] = (_Float16)Ap0[k0 + 32 + o];
                    a3.e[j] = (_Float16)Ap1[k0 + 32 + o];
                    b1.e[j] = (_Float16)Bg[(long)(k0 + 32 + o) * N];
                }
                acc0  = WMMA_F16(acc0,  a0.v, b0.v);
                acc1  = WMMA_F16(acc1,  a1.v, b0.v);
                acc0b = WMMA_F16(acc0b, a2.v, b1.v);
                acc1b = WMMA_F16(acc1b, a3.v, b1.v);
            }
            for (; k0 < Kmain; k0 += 32) {
                V16H a0, a1, b0;
#pragma unroll
                for (int j = 0; j < 16; ++j) {
                    const int o = ((j >> 3) << 4) + (j & 7);
                    a0.e[j] = (_Float16)Ap0[k0 + o];
                    a1.e[j] = (_Float16)Ap1[k0 + o];
                    b0.e[j] = (_Float16)Bg[(long)(k0 + o) * N];
                }
                acc0 = WMMA_F16(acc0, a0.v, b0.v);
                acc1 = WMMA_F16(acc1, a1.v, b0.v);
            }
            if (k0 < K) {    // ragged K tail, branch-free selects
                V16H a0, a1, b0;
#pragma unroll
                for (int j = 0; j < 16; ++j) {
                    const int k  = k0 + ((j >> 3) << 4) + kbase + (j & 7);
                    const int kc = k < K ? k : K - 1;
                    const float av0 = A[(long)(mt      + ml) * K + kc];
                    const float av1 = A[(long)(mt + 16 + ml) * K + kc];
                    const float bv  = Bp[(long)kc * N + nt + ml];
                    a0.e[j] = (_Float16)(k < K ? av0 : 0.0f);
                    a1.e[j] = (_Float16)(k < K ? av1 : 0.0f);
                    b0.e[j] = (_Float16)(k < K ? bv  : 0.0f);
                }
                acc0 = WMMA_F16(acc0, a0.v, b0.v);
                acc1 = WMMA_F16(acc1, a1.v, b0.v);
            }
        }
        acc0 = acc0 + acc0b;
        acc1 = acc1 + acc1b;

        // unguarded epilogue: acc0 -> rows mt.., acc1 -> rows mt+16..
        V8F r0, r1; r0.v = acc0; r1.v = acc1;
        const int n = nt + ml;
#pragma unroll
        for (int i = 0; i < 8; ++i) {
            const int m0 = mt + (half << 3) + i;
            const int m1 = m0 + 16;
            float v0 = r0.e[i], v1 = r1.e[i];
            if (scale)    { v0 *= scale[m0]; v1 *= scale[m1]; }
            if (bias)     { v0 += bias[m0];  v1 += bias[m1];  }
            if (residual) {
                const float* rp = residual + (long)blockIdx.y * strideR + n;
                v0 += rp[(long)m0 * N];
                v1 += rp[(long)m1 * N];
            }
            if (act == 1) { v0 = fmaxf(v0, 0.0f); v1 = fmaxf(v1, 0.0f); }
            else if (act == 2) {
                v0 = fminf(fmaxf(v0, 0.0f), 6.0f);
                v1 = fminf(fmaxf(v1, 0.0f), 6.0f);
            }
            Cp[(long)m0 * N + n] = v0;
            Cp[(long)m1 * N + n] = v1;
        }
    } else {
        // ---- edge path: unconditional clamped loads, cndmask selects ------
        const int m0 = mt + ml, m1 = mt + 16 + ml;
        const int m0C = m0 < M ? m0 : M - 1;
        const int m1C = m1 < M ? m1 : M - 1;
        const int nC  = (nt + ml) < N ? (nt + ml) : N - 1;
        const bool m0OK = m0 < M, m1OK = m1 < M;
        const bool nOK  = (nt + ml) < N;
        const float* __restrict__ Ar0 = A + (long)m0C * K;
        const float* __restrict__ Ar1 = A + (long)m1C * K;
        if (transB) {
            const float* __restrict__ Br = Bp + (long)nC * K;
            for (int k0 = 0; k0 < K; k0 += 32) {
                V16H a0, a1, b0;
#pragma unroll
                for (int j = 0; j < 16; ++j) {
                    const int k  = k0 + ((j >> 3) << 4) + kbase + (j & 7);
                    const int kc = k < K ? k : K - 1;
                    const float av0 = Ar0[kc];   // always in-bounds
                    const float av1 = Ar1[kc];
                    const float bv  = Br[kc];
                    a0.e[j] = (_Float16)((k < K && m0OK) ? av0 : 0.0f);
                    a1.e[j] = (_Float16)((k < K && m1OK) ? av1 : 0.0f);
                    b0.e[j] = (_Float16)((k < K && nOK)  ? bv  : 0.0f);
                }
                acc0 = WMMA_F16(acc0, a0.v, b0.v);
                acc1 = WMMA_F16(acc1, a1.v, b0.v);
            }
        } else {
            const float* __restrict__ Bc = Bp + nC;
            for (int k0 = 0; k0 < K; k0 += 32) {
                V16H a0, a1, b0;
#pragma unroll
                for (int j = 0; j < 16; ++j) {
                    const int k  = k0 + ((j >> 3) << 4) + kbase + (j & 7);
                    const int kc = k < K ? k : K - 1;
                    const float av0 = Ar0[kc];   // always in-bounds
                    const float av1 = Ar1[kc];
                    const float bv  = Bc[(long)kc * N];
                    a0.e[j] = (_Float16)((k < K && m0OK) ? av0 : 0.0f);
                    a1.e[j] = (_Float16)((k < K && m1OK) ? av1 : 0.0f);
                    b0.e[j] = (_Float16)((k < K && nOK)  ? bv  : 0.0f);
                }
                acc0 = WMMA_F16(acc0, a0.v, b0.v);
                acc1 = WMMA_F16(acc1, a1.v, b0.v);
            }
        }
        V8F r0, r1; r0.v = acc0; r1.v = acc1;
        const int n = nt + ml;
        if (n < N) {
#pragma unroll
            for (int i = 0; i < 8; ++i) {
#pragma unroll
                for (int hRow = 0; hRow < 2; ++hRow) {
                    const int m = mt + (hRow << 4) + (half << 3) + i;
                    if (m < M) {
                        float v = hRow ? r1.e[i] : r0.e[i];
                        if (scale)    v *= scale[m];
                        if (bias)     v += bias[m];
                        if (residual) v += residual[(long)blockIdx.y * strideR +
                                                    (long)m * N + n];
                        if (act == 1)      v = fmaxf(v, 0.0f);
                        else if (act == 2) v = fminf(fmaxf(v, 0.0f), 6.0f);
                        Cp[(long)m * N + n] = v;
                    }
                }
            }
        }
    }
}

// ---------------------------------------------------------------------------
// Direct 3x3 conv (pad=1), fp32 VALU.  Used for comp (12->3) and stem (3->32).
// ---------------------------------------------------------------------------
__global__ void conv3x3_kernel(const float* __restrict__ in,
                               const float* __restrict__ w,
                               const float* __restrict__ scale,
                               const float* __restrict__ bias,
                               float* __restrict__ out,
                               int NB, int Cin, int Cout,
                               int Hin, int Win, int Hout, int Wout,
                               int stride, int act)
{
    long idx = (long)blockIdx.x * blockDim.x + threadIdx.x;
    long total = (long)NB * Cout * Hout * Wout;
    if (idx >= total) return;
    int wo = (int)(idx % Wout); long t = idx / Wout;
    int ho = (int)(t % Hout);   t /= Hout;
    int co = (int)(t % Cout);   int b = (int)(t / Cout);

    const float* ip = in + (long)b * Cin * Hin * Win;
    const float* wp = w  + (long)co * Cin * 9;
    const int hi0 = ho * stride - 1, wi0 = wo * stride - 1;
    float s = 0.0f;
    for (int ci = 0; ci < Cin; ++ci) {
        const float* ipc = ip + (long)ci * Hin * Win;
        const float* wpc = wp + ci * 9;
#pragma unroll
        for (int kh = 0; kh < 3; ++kh) {
            int hi = hi0 + kh;
            if ((unsigned)hi >= (unsigned)Hin) continue;
#pragma unroll
            for (int kw = 0; kw < 3; ++kw) {
                int wi = wi0 + kw;
                if ((unsigned)wi >= (unsigned)Win) continue;
                s += ipc[(long)hi * Win + wi] * wpc[kh * 3 + kw];
            }
        }
    }
    if (scale) s *= scale[co];
    if (bias)  s += bias[co];
    if (act == 1)      s = fmaxf(s, 0.0f);
    else if (act == 2) s = fminf(fmaxf(s, 0.0f), 6.0f);
    out[idx] = s;
}

// ---------------------------------------------------------------------------
// Depthwise 3x3 conv (pad=1) + BN + relu6, fp32 VALU (memory bound).
// ---------------------------------------------------------------------------
__global__ void dwconv3x3_kernel(const float* __restrict__ in,
                                 const float* __restrict__ w,
                                 const float* __restrict__ scale,
                                 const float* __restrict__ bias,
                                 float* __restrict__ out,
                                 int NB, int C, int Hin, int Win,
                                 int Hout, int Wout, int stride)
{
    long idx = (long)blockIdx.x * blockDim.x + threadIdx.x;
    long total = (long)NB * C * Hout * Wout;
    if (idx >= total) return;
    int wo = (int)(idx % Wout); long t = idx / Wout;
    int ho = (int)(t % Hout);   t /= Hout;
    int c  = (int)(t % C);      int b = (int)(t / C);

    const float* ip = in + ((long)b * C + c) * Hin * Win;
    const float* wp = w + c * 9;
    const int hi0 = ho * stride - 1, wi0 = wo * stride - 1;
    float s = 0.0f;
#pragma unroll
    for (int kh = 0; kh < 3; ++kh) {
        int hi = hi0 + kh;
        if ((unsigned)hi >= (unsigned)Hin) continue;
#pragma unroll
        for (int kw = 0; kw < 3; ++kw) {
            int wi = wi0 + kw;
            if ((unsigned)wi >= (unsigned)Win) continue;
            s += ip[(long)hi * Win + wi] * wp[kh * 3 + kw];
        }
    }
    s = s * scale[c] + bias[c];
    out[idx] = fminf(fmaxf(s, 0.0f), 6.0f);
}

// ---------------------------------------------------------------------------
// Global average pool over HW.  in: (NB, C, HW) -> out: (NB, C)
// ---------------------------------------------------------------------------
__global__ void avgpool_kernel(const float* __restrict__ in,
                               float* __restrict__ out, int NB, int C, int HW)
{
    int idx = blockIdx.x * blockDim.x + threadIdx.x;
    if (idx >= NB * C) return;
    const float* p = in + (long)idx * HW;
    float s = 0.0f;
    for (int i = 0; i < HW; ++i) s += p[i];
    out[idx] = s / (float)HW;
}

// ---------------------------------------------------------------------------
// Initialize LSTM rollout state: ct = broadcast(c0), st0 = [0,0,0,v],
// cs row 1000 = current_v (cs is stored transposed, 1001 x 32).
// ---------------------------------------------------------------------------
__global__ void init_state_kernel(const float* __restrict__ v,
                                  const float* __restrict__ c0,
                                  float* __restrict__ ct,
                                  float* __restrict__ st,
                                  float* __restrict__ cs)
{
    int idx = blockIdx.x * blockDim.x + threadIdx.x;
    if (idx < 4096) {                      // ct (128 x 32), column-batch
        int h = idx >> 5, b = idx & 31;
        ct[h * 32 + b] = c0[h];
    }
    if (idx < 32) {
        st[idx * 4 + 0] = 0.0f;
        st[idx * 4 + 1] = 0.0f;
        st[idx * 4 + 2] = 0.0f;
        st[idx * 4 + 3] = v[idx];
        cs[1000 * 32 + idx] = v[idx];
    }
}

// ---------------------------------------------------------------------------
// One LSTM + kinematic-bicycle rollout step. Single 256-thread block (8 waves).
// ht/ct stored transposed (128 x 32) so batch index is the fast axis.
// ---------------------------------------------------------------------------
__global__ void __launch_bounds__(256)
lstm_step_kernel(const float* __restrict__ htIn, float* __restrict__ htOut,
                 float* __restrict__ ct, float* __restrict__ st,
                 const float* __restrict__ cs,
                 const float* __restrict__ out_w, const float* __restrict__ out_b,
                 const float* __restrict__ emb_w, const float* __restrict__ emb_b,
                 const float* __restrict__ Wih, const float* __restrict__ Whh,
                 const float* __restrict__ bih, const float* __restrict__ bhh,
                 float* __restrict__ traj, int stepIdx)
{
    __shared__ float sSteer[32];
    __shared__ float sAcc[32];
    __shared__ float sE[32 * 64];
    const int t = threadIdx.x;

    // --- act = [ht, cs] @ out_w.T + out_b ;  steer/acc clipping ---
    if (t < 64) {
        const int b = t >> 1, j = t & 1;
        const float* wr = out_w + j * 1129;
        float s = out_b[j];
        for (int i = 0; i < 128; ++i)  s += wr[i] * htIn[i * 32 + b];
        for (int k = 0; k < 1001; ++k) s += wr[128 + k] * cs[k * 32 + b];
        if (j == 0) sSteer[b] = fminf(fmaxf(s, -0.52f), 0.52f);
        else        sAcc[b]   = fminf(fmaxf(s, -6.0f),  2.0f);
    }
    __syncthreads();

    // --- kinematic bicycle update + trajectory emit ---
    if (t < 32) {
        const int b = t;
        float x = st[b * 4 + 0], y = st[b * 4 + 1];
        float th = st[b * 4 + 2], v = st[b * 4 + 3];
        x  += v * cosf(th) * 0.2f;
        y  += v * sinf(th) * 0.2f;
        th += v * tanf(sSteer[b]) * (0.2f / 2.8f);
        v   = fmaxf(v + sAcc[b] * 0.2f, 0.0f);
        st[b * 4 + 0] = x; st[b * 4 + 1] = y;
        st[b * 4 + 2] = th; st[b * 4 + 3] = v;
        float* o = traj + b * 40 + stepIdx * 4;
        o[0] = x; o[1] = y; o[2] = th; o[3] = v;
    }
    __syncthreads();

    // --- e = relu([steer,acc] @ emb_w.T + emb_b), 32 x 64 in LDS ---
    for (int idx = t; idx < 32 * 64; idx += 256) {
        const int b = idx >> 6, j = idx & 63;
        float e = sSteer[b] * emb_w[j * 2 + 0] + sAcc[b] * emb_w[j * 2 + 1]
                + emb_b[j];
        sE[idx] = fmaxf(e, 0.0f);
    }
    __syncthreads();

    // --- gates + cell update: 32 batches x 128 hidden ---
    for (int idx = t; idx < 32 * 128; idx += 256) {
        const int b = idx & 31, h = idx >> 5;
        float g4[4];
#pragma unroll
        for (int gq = 0; gq < 4; ++gq) {
            const int row = gq * 128 + h;
            float s = bih[row] + bhh[row];
            const float* wi = Wih + row * 64;
            const float* wh = Whh + row * 128;
            const float* eb = sE + b * 64;
            for (int k = 0; k < 64; ++k)  s += wi[k] * eb[k];
            for (int k = 0; k < 128; ++k) s += wh[k] * htIn[k * 32 + b];
            g4[gq] = s;
        }
        const float gi = 1.0f / (1.0f + expf(-g4[0]));
        const float gf = 1.0f / (1.0f + expf(-g4[1]));
        const float gg = tanhf(g4[2]);
        const float go = 1.0f / (1.0f + expf(-g4[3]));
        const float c2 = gf * ct[h * 32 + b] + gi * gg;
        ct[h * 32 + b] = c2;
        htOut[h * 32 + b] = go * tanhf(c2);
    }
}

// ---------------------------------------------------------------------------
// Host orchestration
// ---------------------------------------------------------------------------
extern "C" void kernel_launch(void* const* d_in, const int* in_sizes, int n_in,
                              void* d_out, int out_size, void* d_ws, size_t ws_size,
                              hipStream_t stream)
{
    (void)in_sizes; (void)n_in; (void)out_size; (void)ws_size;
    const int NB = 32;
    static const int CFG[7][4] = {
        {1, 16, 1, 1}, {6, 24, 2, 2}, {6, 32, 3, 2}, {6, 64, 4, 2},
        {6, 96, 3, 1}, {6, 160, 3, 2}, {6, 320, 1, 1}};

    // ---- walk input pointers in setup_inputs() dict-insertion order ----
    int ix = 0;
    auto nxt = [&]() { return (const float*)d_in[ix++]; };
    const float* img    = nxt();           // (32,12,224,224)
    const float* cur_v  = nxt();           // (32,1)
    const float* comp_w = nxt(); const float* comp_b = nxt();
    const float* h0_w1  = nxt(); const float* h0_b1  = nxt();
    const float* h0_w2  = nxt(); const float* h0_b2  = nxt();
    const float* c0     = nxt();
    const float* emb_w  = nxt(); const float* emb_b  = nxt();
    const float* out_w  = nxt(); const float* out_b  = nxt();
    const float* Wih    = nxt(); const float* Whh    = nxt();
    const float* bih    = nxt(); const float* bhh    = nxt();
    const float* stem_w = nxt(); const float* stem_s = nxt();
    const float* stem_b = nxt();
    struct Blk { const float *ew, *es, *eb, *dw, *ds, *db, *pw, *ps, *pb; };
    Blk blk[17];
    {
        int bi = 0;
        for (int g = 0; g < 7; ++g)
            for (int i = 0; i < CFG[g][2]; ++i) {
                Blk& B = blk[bi++];
                if (CFG[g][0] != 1) { B.ew = nxt(); B.es = nxt(); B.eb = nxt(); }
                else                { B.ew = nullptr; B.es = nullptr; B.eb = nullptr; }
                B.dw = nxt(); B.ds = nxt(); B.db = nxt();
                B.pw = nxt(); B.ps = nxt(); B.pb = nxt();
            }
    }
    const float* head_w = nxt(); const float* head_s = nxt();
    const float* head_b = nxt();
    const float* fc_w   = nxt(); const float* fc_b   = nxt();

    // ---- workspace carve (floats) ----
    float* ws = (float*)d_ws;
    size_t off = 0;
    auto carve = [&](size_t n) { float* p = ws + off; off += n; return p; };
    float* xA   = carve(12845056);   // max x: 32ch @ 112^2, batch 32
    float* xB   = carve(12845056);
    float* bufH = carve(38535168);   // max hidden: 96ch @ 112^2, batch 32
    float* bufD = carve(12845056);   // 32ch @ 112^2, batch 32; max dw-out is 144ch @ 56^2
    float* pooled = carve(32 * 1280);
    float* cs     = carve(1001 * 32);  // [emb^T ; v^T], 1001 x 32
    float* hbuf   = carve(512 * 32);
    float* htA    = carve(128 * 32);
    float* htB    = carve(128 * 32);
    float* ct     = carve(128 * 32);
    float* st     = carve(32 * 4);

    auto gemm = [&](const float* A, const float* B, float* C,
                    int M, int N, int K, int transB,
                    const float* scale, const float* bias,
                    const float* residual, int act,
                    long sB, long sC, long sR, int batch) {
        const int tiles = ((M + 31) / 32) * ((N + 15) / 16);
        dim3 grid((tiles + 7) / 8, batch);
        gemm_wmma_kernel<<<grid, 256, 0, stream>>>(
            A, B, C, M, N, K, transB, scale, bias, residual, act, sB, sC, sR);
    };

    // ---- comp conv: 12 -> 3, 224x224, bias only ----
    {
        long total = (long)NB * 3 * 224 * 224;
        conv3x3_kernel<<<(total + 255) / 256, 256, 0, stream>>>(
            img, comp_w, nullptr, comp_b, xA, NB, 12, 3, 224, 224, 224, 224, 1, 0);
    }
    // ---- stem: 3 -> 32, stride 2, BN + relu6 ----
    {
        long total = (long)NB * 32 * 112 * 112;
        conv3x3_kernel<<<(total + 255) / 256, 256, 0, stream>>>(
            xA, stem_w, stem_s, stem_b, xB, NB, 3, 32, 224, 224, 112, 112, 2, 2);
    }

    // ---- MobileNetV2 inverted-residual blocks ----
    float* x = xB; float* xAlt = xA;
    int H = 112, W = 112, in_c = 32, bi = 0;
    for (int g = 0; g < 7; ++g) {
        const int t = CFG[g][0], c = CFG[g][1], n = CFG[g][2], s0 = CFG[g][3];
        for (int i = 0; i < n; ++i) {
            const int stride = (i == 0) ? s0 : 1;
            const Blk& B = blk[bi++];
            const int hid = in_c * t;
            const int Ho = H / stride, Wo = W / stride;

            const float* hin;
            if (t != 1) {   // expand 1x1 conv: WMMA GEMM, relu6
                gemm(B.ew, x, bufH, hid, H * W, in_c, 0, B.es, B.eb, nullptr, 2,
                     (long)in_c * H * W, (long)hid * H * W, 0, NB);
                hin = bufH;
            } else hin = x;

            {               // depthwise 3x3 + BN + relu6
                long total = (long)NB * hid * Ho * Wo;
                dwconv3x3_kernel<<<(total + 255) / 256, 256, 0, stream>>>(
                    hin, B.dw, B.ds, B.db, bufD, NB, hid, H, W, Ho, Wo, stride);
            }

            // project 1x1 conv: WMMA GEMM, linear, residual fused in epilogue
            const float* res = (stride == 1 && in_c == c) ? x : nullptr;
            gemm(B.pw, bufD, xAlt, c, Ho * Wo, hid, 0, B.ps, B.pb, res, 0,
                 (long)hid * Ho * Wo, (long)c * Ho * Wo, (long)c * Ho * Wo, NB);

            float* tmp = x; x = xAlt; xAlt = tmp;
            in_c = c; H = Ho; W = Wo;
        }
    }

    // ---- head 1x1 conv 320 -> 1280 @ 7x7, relu6 (WMMA) ----
    gemm(head_w, x, bufH, 1280, H * W, 320, 0, head_s, head_b, nullptr, 2,
         (long)320 * H * W, (long)1280 * H * W, 0, NB);
    // ---- global average pool -> (32, 1280) ----
    avgpool_kernel<<<(32 * 1280 + 255) / 256, 256, 0, stream>>>(
        bufH, pooled, NB, 1280, H * W);
    // ---- fc: emb^T (1000 x 32) = fc_w (1000x1280) x pooled^T (WMMA, transB) ----
    gemm(fc_w, pooled, cs, 1000, 32, 1280, 1, nullptr, fc_b, nullptr, 0,
         0, 0, 0, 1);
    // ---- ct/st init + cs row 1000 = current_v ----
    init_state_kernel<<<16, 256, 0, stream>>>(cur_v, c0, ct, st, cs);
    // ---- h0 MLP: 1001 -> 512 -> 128, relu (WMMA) ----
    gemm(h0_w1, cs, hbuf, 512, 32, 1001, 0, nullptr, h0_b1, nullptr, 1, 0, 0, 0, 1);
    gemm(h0_w2, hbuf, htA, 128, 32, 512, 0, nullptr, h0_b2, nullptr, 1, 0, 0, 0, 1);

    // ---- 10-step LSTM + kinematic rollout ----
    float* hi = htA; float* ho = htB;
    for (int step = 0; step < 10; ++step) {
        lstm_step_kernel<<<1, 256, 0, stream>>>(
            hi, ho, ct, st, cs, out_w, out_b, emb_w, emb_b,
            Wih, Whh, bih, bhh, (float*)d_out, step);
        float* tmp = hi; hi = ho; ho = tmp;
    }
}
